// StandardPartitionAttention_77996606096041
// MI455X (gfx1250) — matrix operand order repaired and checked by reading, not compiled, not run on hardware
//
#include <hip/hip_runtime.h>
#include <hip/hip_bf16.h>

// ---------------------------------------------------------------------------
// Swin-style windowed attention for MI455X (gfx1250, wave32, WMMA).
// All GEMMs: v_wmma_f32_16x16x32_f16 (f16 inputs, f32 accumulate).
// Per-window workgroups: 256 threads = 8 wave32s, __launch_bounds__(256,1)
// so the allocator can hold 64-VGPR accumulators + 64-VGPR B staging without
// scratch spills. B-tiles pre-packed kt-major: one k-step's 8 tiles = one
// contiguous 8KB block -> single s_clause of 16 global_load_b128, WMMAs run
// under partial s_wait_loadcnt.
// MLP keeps its full 64x1024 ReLU intermediate in LDS (CDNA5: 320KB/WGP).
// ---------------------------------------------------------------------------

typedef __attribute__((ext_vector_type(16))) _Float16 v16h;
typedef __attribute__((ext_vector_type(8)))  _Float16 v8h;
typedef __attribute__((ext_vector_type(8)))  float    v8f;

#define NTOK    64          // tokens per 8x8 window
#define DMODEL  256
#define NWIN    1024
#define LDSTR   264         // LDS row stride in halves (16B-aligned, staggered)
#define MIDSTR  1032        // LDS row stride for 1024-wide MLP intermediate
#define SCALE_F 0.17677669529663687f   // 32^-0.5

__device__ __forceinline__ v8f zero8() {
  v8f z;
#pragma unroll
  for (int i = 0; i < 8; ++i) z[i] = 0.0f;
  return z;
}

__device__ __forceinline__ v8f wmma16(v16h a, v16h b, v8f c) {
  // D = A(16x32 f16) * B(32x16 f16) + C(16x16 f32)
  return __builtin_amdgcn_wmma_f32_16x16x32_f16(false, a, false, b, (short)0, c,
                                                false, false);
}

__device__ __forceinline__ v16h join16(v8h lo, v8h hi) {
  v16h r;
#pragma unroll
  for (int i = 0; i < 8; ++i) { r[i] = lo[i]; r[i + 8] = hi[i]; }
  return r;
}

// A fragment (16x32, f16) from row-major tile. Layout:
//   M = lane&15 ; element e -> K = e + 16*(lane>>4)
// => two contiguous 16B loads at K-offsets 16*(lane>>4) and +8.
__device__ __forceinline__ v16h frag_a(const _Float16* tile, int stride, int lane) {
  const _Float16* p = tile + (lane & 15) * stride + ((lane >> 4) << 4);
  v8h lo = *(const v8h*)p;
  v8h hi = *(const v8h*)(p + 8);
  return join16(lo, hi);
}

// One wave computes 8 16x16 C-tiles of a 64xN GEMM (A in LDS, B pre-packed).
// Packed B tile index = kt*NT_B + nt ; each tile = 32 lanes * 16 halves (1KB).
// Batch-load all 8 B fragments, then issue the 8 WMMAs (partial waits).
__device__ __forceinline__ void gemm_wave(const _Float16* ldsA, int lds_stride,
                                          const _Float16* __restrict__ pB, int NT_B,
                                          int ktBegin, int ktCount, int ntBase,
                                          int mt, int lane, v8f acc[8]) {
  const _Float16* lanep = pB + (size_t)lane * 16;
  for (int kt = 0; kt < ktCount; ++kt) {
    v16h a = frag_a(ldsA + mt * 16 * lds_stride + kt * 32, lds_stride, lane);
    const _Float16* bbase =
        lanep + ((size_t)(ktBegin + kt) * NT_B + ntBase) * 512;
    v16h b[8];
#pragma unroll
    for (int t = 0; t < 8; ++t)
      b[t] = *(const v16h*)(bbase + (size_t)t * 512);
#pragma unroll
    for (int t = 0; t < 8; ++t)
      acc[t] = wmma16(a, b[t], acc[t]);
  }
}

// LayerNorm over D=256 per token -> f16 into LDS A (stride LDSTR).
// 256 threads: 64 rows x 4 partials.
__device__ __forceinline__ void layernorm_to_lds(const float* __restrict__ Tw,
                                                 const float* __restrict__ g,
                                                 const float* __restrict__ b,
                                                 _Float16* A, float* red) {
  int tid = threadIdx.x;
  int r = tid >> 2, part = tid & 3;
  const float* p = Tw + r * 256 + part * 64;
  float s = 0.f, q = 0.f;
#pragma unroll 8
  for (int i = 0; i < 64; ++i) { float v = p[i]; s += v; q += v * v; }
  red[2 * tid] = s; red[2 * tid + 1] = q;
  __syncthreads();
  if (part == 0) {
    float ss = red[2*tid] + red[2*tid+2] + red[2*tid+4] + red[2*tid+6];
    float qq = red[2*tid+1] + red[2*tid+3] + red[2*tid+5] + red[2*tid+7];
    float mean = ss * (1.0f / 256.0f);
    float var  = qq * (1.0f / 256.0f) - mean * mean;
    red[2 * tid]     = mean;
    red[2 * tid + 1] = rsqrtf(var + 1e-5f);
  }
  __syncthreads();
  float mean = red[8 * r], rstd = red[8 * r + 1];
#pragma unroll 8
  for (int i = 0; i < 64; ++i) {
    int c = part * 64 + i;
    A[r * LDSTR + c] = (_Float16)((p[i] - mean) * rstd * g[c] + b[c]);
  }
  __syncthreads();
}

// ---------------------------------------------------------------------------
// Kernel: pack f32 weight [K][N] into f16 B-fragment tiles, kt-major order.
// B layout (32x16 tile): N = lane&15, element e -> K = e + 16*(lane>>4).
// ---------------------------------------------------------------------------
__global__ void pack_w_k(const float* __restrict__ src, _Float16* __restrict__ dst,
                         int K, int N) {
  int tile = blockIdx.x * 8 + (threadIdx.x >> 5);
  int lane = threadIdx.x & 31;
  int KT = K >> 5, NT = N >> 4;
  if (tile >= KT * NT) return;
  int kt = tile / NT, nt = tile % NT;
  int n = nt * 16 + (lane & 15);
  int kbase = kt * 32 + 16 * (lane >> 4);
  _Float16* out = dst + (size_t)tile * 512 + lane * 16;
#pragma unroll
  for (int e = 0; e < 16; ++e)
    out[e] = (_Float16)src[(size_t)(kbase + e) * N + n];
}

// ---------------------------------------------------------------------------
// Kernel: per-layer relative-position bias + gaussian decay, [8][64][64] f32.
// ---------------------------------------------------------------------------
__global__ void biaspos_k(const float* __restrict__ btab,   // [225][8]
                          const float* __restrict__ sita,   // [8]
                          float* __restrict__ BP) {
  int idx = blockIdx.x * 256 + threadIdx.x;                 // 8*4096
  int h = idx >> 12, i = (idx >> 6) & 63, j = idx & 63;
  int yi = i >> 3, xi = i & 7, yj = j >> 3, xj = j & 7;
  int rpi = ((yi - yj) + 7) * 15 + ((xi - xj) + 7);
  float dy = (float)(yi - yj) * 0.125f, dx = (float)(xi - xj) * 0.125f;
  float dis = dy * dy + dx * dx;
  float st = sita[h];
  float factor = 1.0f / (2.0f * st * st + 1e-10f);
  BP[idx] = btab[rpi * 8 + h] + 0.01f * __expf(-factor * dis);
}

// ---------------------------------------------------------------------------
// Kernel: window partition + patch embedding:  T[w][i][:] = x_win @ Wp + bp
// ---------------------------------------------------------------------------
__global__ void __launch_bounds__(256, 1)
patch_embed_k(const float* __restrict__ x,
              const _Float16* __restrict__ pW,
              const float* __restrict__ bp,
              float* __restrict__ T) {
  __shared__ _Float16 A[NTOK * LDSTR];
  int w = blockIdx.x;
  int b = w >> 8, wy = (w >> 4) & 15, wx = w & 15;
  int tid = threadIdx.x, lane = tid & 31, wave = tid >> 5;
  int i = tid & 63;
  int r = wy * 8 + (i >> 3), c = wx * 8 + (i & 7);
  const float* xp = x + (((size_t)b * 256) * 128 + r) * 128 + c;
  for (int k = tid >> 6; k < 256; k += 4)
    A[i * LDSTR + k] = (_Float16)xp[(size_t)k * 16384];
  __syncthreads();
  int mt = wave & 3, ntBase = (wave >> 2) * 8;
  v8f acc[8];
#pragma unroll
  for (int t = 0; t < 8; ++t) acc[t] = zero8();
  gemm_wave(A, LDSTR, pW, 16, 0, 8, ntBase, mt, lane, acc);
  float* Tw = T + (size_t)w * NTOK * DMODEL;
  int rAdd = (lane >> 4) << 3;
#pragma unroll
  for (int t = 0; t < 8; ++t) {
    int col = (ntBase + t) * 16 + (lane & 15);
    float bb = bp[col];
#pragma unroll
    for (int jj = 0; jj < 8; ++jj) {
      int row = mt * 16 + jj + rAdd;
      Tw[row * DMODEL + col] = acc[t][jj] + bb;
    }
  }
}

// ---------------------------------------------------------------------------
// Kernel: LN1 + QKV projection. q prescaled by dh^-0.5. v stored transposed.
// Qh,Kh: [w][h][64][32] f16 ; Vt: [w][h][32][64] f16
// ---------------------------------------------------------------------------
__global__ void __launch_bounds__(256, 1)
ln_qkv_k(const float* __restrict__ T, const float* __restrict__ g,
         const float* __restrict__ bt,
         const _Float16* __restrict__ pW,
         _Float16* __restrict__ Qh, _Float16* __restrict__ Kh,
         _Float16* __restrict__ Vt) {
  __shared__ _Float16 A[NTOK * LDSTR];
  __shared__ float red[512];
  int w = blockIdx.x;
  int tid = threadIdx.x, lane = tid & 31, wave = tid >> 5;
  layernorm_to_lds(T + (size_t)w * 16384, g, bt, A, red);
  int mt = wave & 3, ntBase = (wave >> 2) * 8;
  int rAdd = (lane >> 4) << 3;
  for (int chunk = 0; chunk < 3; ++chunk) {        // q, k, v (N-cols of Wqkv)
    v8f acc[8];
#pragma unroll
    for (int t = 0; t < 8; ++t) acc[t] = zero8();
    gemm_wave(A, LDSTR, pW, 48, 0, 8, chunk * 16 + ntBase, mt, lane, acc);
#pragma unroll
    for (int t = 0; t < 8; ++t) {
      int col = (ntBase + t) * 16 + (lane & 15);
      int head = col >> 5, d = col & 31;
      size_t hb = ((size_t)w * 8 + head);
#pragma unroll
      for (int jj = 0; jj < 8; ++jj) {
        int row = mt * 16 + jj + rAdd;
        float v = acc[t][jj];
        if (chunk == 0)      Qh[(hb * 64 + row) * 32 + d] = (_Float16)(v * SCALE_F);
        else if (chunk == 1) Kh[(hb * 64 + row) * 32 + d] = (_Float16)v;
        else                 Vt[(hb * 32 + d) * 64 + row] = (_Float16)v;
      }
    }
  }
}

// ---------------------------------------------------------------------------
// Kernel: attention. One wave per head. dots = q@k^T (one WMMA K-step, dh=32),
// dual softmax (unbiased -> attns output; biased -> attn), attn@v via LDS.
// ---------------------------------------------------------------------------
__global__ void __launch_bounds__(256, 1)
attn_k(const _Float16* __restrict__ Qh,
       const _Float16* __restrict__ Kh,
       const _Float16* __restrict__ Vt,
       const float* __restrict__ BP,       // [8][64][64]
       float* __restrict__ attn0,          // [w][h][64][64] f32
       _Float16* __restrict__ O) {         // [w][64][256] f16
  extern __shared__ char dyn_smem[];
  _Float16* AT = (_Float16*)dyn_smem;                      // [8][64][72]
  int w = blockIdx.x;
  int head = threadIdx.x >> 5, lane = threadIdx.x & 31;
  const _Float16* q  = Qh + ((size_t)w * 8 + head) * 2048;
  const _Float16* kk = Kh + ((size_t)w * 8 + head) * 2048;
  const _Float16* vt = Vt + ((size_t)w * 8 + head) * 2048;
  int cbase = lane & 15, rAdd = (lane >> 4) << 3;

  v16h af[4], bf[4];
#pragma unroll
  for (int mt = 0; mt < 4; ++mt) {
    const _Float16* p = q + (mt * 16 + cbase) * 32 + ((lane >> 4) << 4);
    af[mt] = join16(*(const v8h*)p, *(const v8h*)(p + 8));
  }
#pragma unroll
  for (int nt = 0; nt < 4; ++nt)
    bf[nt] = *(const v16h*)(kk + (nt * 16 + cbase) * 32 + ((lane >> 4) << 4));

  v8f d0[4][4];
#pragma unroll
  for (int mt = 0; mt < 4; ++mt)
#pragma unroll
    for (int nt = 0; nt < 4; ++nt) d0[mt][nt] = zero8();
#pragma unroll
  for (int mt = 0; mt < 4; ++mt)
#pragma unroll
    for (int nt = 0; nt < 4; ++nt)
      d0[mt][nt] = wmma16(af[mt], bf[nt], d0[mt][nt]);

  const float* bpp = BP + head * 4096;
  float* a0 = attn0 + ((size_t)w * 8 + head) * 4096;
  _Float16* atl = AT + head * 64 * 72;
#pragma unroll
  for (int mt = 0; mt < 4; ++mt) {
#pragma unroll
    for (int jj = 0; jj < 8; ++jj) {
      int row = mt * 16 + jj + rAdd;
      float v[4], bb[4];
      float mx = -3.0e38f, mx2 = -3.0e38f;
#pragma unroll
      for (int nt = 0; nt < 4; ++nt) {
        v[nt]  = d0[mt][nt][jj];
        bb[nt] = v[nt] + bpp[row * 64 + nt * 16 + cbase];
        mx  = fmaxf(mx, v[nt]);
        mx2 = fmaxf(mx2, bb[nt]);
      }
      for (int off = 8; off; off >>= 1) {       // 16-lane row reductions
        mx  = fmaxf(mx,  __shfl_xor(mx,  off, 32));
        mx2 = fmaxf(mx2, __shfl_xor(mx2, off, 32));
      }
      float s = 0.f, s2 = 0.f;
#pragma unroll
      for (int nt = 0; nt < 4; ++nt) {
        v[nt]  = __expf(v[nt]  - mx);  s  += v[nt];
        bb[nt] = __expf(bb[nt] - mx2); s2 += bb[nt];
      }
      for (int off = 8; off; off >>= 1) {
        s  += __shfl_xor(s,  off, 32);
        s2 += __shfl_xor(s2, off, 32);
      }
      float is = 1.f / s, is2 = 1.f / s2;
#pragma unroll
      for (int nt = 0; nt < 4; ++nt) {
        a0[row * 64 + nt * 16 + cbase]  = v[nt] * is;
        atl[row * 72 + nt * 16 + cbase] = (_Float16)(bb[nt] * is2);
      }
    }
  }
  __syncthreads();

  // o = attn @ v  (M=64, N=32, K=64)
  v8f oa[4][2];
#pragma unroll
  for (int mt = 0; mt < 4; ++mt)
#pragma unroll
    for (int nt = 0; nt < 2; ++nt) oa[mt][nt] = zero8();
  for (int kt = 0; kt < 2; ++kt) {
    v16h bv[2];
#pragma unroll
    for (int nt = 0; nt < 2; ++nt)
      bv[nt] = *(const v16h*)(vt + (nt * 16 + cbase) * 64 + kt * 32 +
                              ((lane >> 4) << 4));
#pragma unroll
    for (int mt = 0; mt < 4; ++mt) {
      v16h a = frag_a(atl + mt * 16 * 72 + kt * 32, 72, lane);
#pragma unroll
      for (int nt = 0; nt < 2; ++nt) oa[mt][nt] = wmma16(a, bv[nt], oa[mt][nt]);
    }
  }
  _Float16* ow = O + (size_t)w * 16384 + head * 32;
#pragma unroll
  for (int mt = 0; mt < 4; ++mt)
#pragma unroll
    for (int nt = 0; nt < 2; ++nt)
#pragma unroll
      for (int jj = 0; jj < 8; ++jj) {
        int row = mt * 16 + jj + rAdd;
        ow[row * 256 + nt * 16 + cbase] = (_Float16)oa[mt][nt][jj];
      }
}

// ---------------------------------------------------------------------------
// Kernel: output projection + residual: T += O @ Wout + bout
// ---------------------------------------------------------------------------
__global__ void __launch_bounds__(256, 1)
outproj_k(const _Float16* __restrict__ O,
          const _Float16* __restrict__ pW,
          const float* __restrict__ bo, float* __restrict__ T) {
  __shared__ _Float16 A[NTOK * LDSTR];
  int w = blockIdx.x, tid = threadIdx.x, lane = tid & 31, wave = tid >> 5;
  {
    int r = tid >> 2, part = tid & 3;
    const v8h* src = (const v8h*)(O + (size_t)w * 16384 + r * 256 + part * 64);
    v8h* dst = (v8h*)(A + r * LDSTR + part * 64);
#pragma unroll
    for (int i = 0; i < 8; ++i) dst[i] = src[i];
  }
  __syncthreads();
  int mt = wave & 3, ntBase = (wave >> 2) * 8;
  v8f acc[8];
#pragma unroll
  for (int t = 0; t < 8; ++t) acc[t] = zero8();
  gemm_wave(A, LDSTR, pW, 16, 0, 8, ntBase, mt, lane, acc);
  float* Tw = T + (size_t)w * 16384;
  int rAdd = (lane >> 4) << 3;
#pragma unroll
  for (int t = 0; t < 8; ++t) {
    int col = (ntBase + t) * 16 + (lane & 15);
    float bb = bo[col];
#pragma unroll
    for (int jj = 0; jj < 8; ++jj) {
      int row = mt * 16 + jj + rAdd;
      Tw[row * 256 + col] += acc[t][jj] + bb;
    }
  }
}

// ---------------------------------------------------------------------------
// Kernel: LN2 + fused MLP (+residual).  T += relu(LN2(T)@W1+b1)@W2 + b2
// Full 64x1024 ReLU intermediate lives in LDS (132KB; CDNA5 has 320KB/WGP),
// so only one 64-VGPR accumulator is live in each GEMM phase (no spills).
// ---------------------------------------------------------------------------
__global__ void __launch_bounds__(256, 1)
ln2_mlp_k(float* __restrict__ T, const float* __restrict__ g,
          const float* __restrict__ bt,
          const _Float16* __restrict__ pW1,
          const float* __restrict__ b1,
          const _Float16* __restrict__ pW2,
          const float* __restrict__ b2) {
  extern __shared__ char dyn_smem[];
  _Float16* A   = (_Float16*)dyn_smem;                              // 64*264 f16
  _Float16* MID = (_Float16*)(dyn_smem + NTOK * LDSTR * 2);         // 64*1032 f16
  float*    red = (float*)(dyn_smem + NTOK * LDSTR * 2 + NTOK * MIDSTR * 2);
  int w = blockIdx.x, tid = threadIdx.x, lane = tid & 31, wave = tid >> 5;
  float* Tw = T + (size_t)w * 16384;
  layernorm_to_lds(Tw, g, bt, A, red);
  int mt = wave & 3, ntBase = (wave >> 2) * 8;
  int rAdd = (lane >> 4) << 3;

  // phase 1: MID = relu(LN2(T) @ W1 + b1), 4 chunks of 256 output cols
  for (int kc = 0; kc < 4; ++kc) {
    v8f macc[8];
#pragma unroll
    for (int t = 0; t < 8; ++t) macc[t] = zero8();
    gemm_wave(A, LDSTR, pW1, 64, 0, 8, kc * 16 + ntBase, mt, lane, macc);
#pragma unroll
    for (int t = 0; t < 8; ++t) {
      int col = (ntBase + t) * 16 + (lane & 15);
      float b1v = b1[kc * 256 + col];
#pragma unroll
      for (int jj = 0; jj < 8; ++jj) {
        int row = mt * 16 + jj + rAdd;
        float v = macc[t][jj] + b1v;
        MID[row * MIDSTR + kc * 256 + col] = (_Float16)(v > 0.f ? v : 0.f);
      }
    }
  }
  __syncthreads();

  // phase 2: T += MID @ W2 + b2 (single 32-k-step pass)
  v8f oacc[8];
#pragma unroll
  for (int t = 0; t < 8; ++t) oacc[t] = zero8();
  gemm_wave(MID, MIDSTR, pW2, 16, 0, 32, ntBase, mt, lane, oacc);
#pragma unroll
  for (int t = 0; t < 8; ++t) {
    int col = (ntBase + t) * 16 + (lane & 15);
    float b2v = b2[col];
#pragma unroll
    for (int jj = 0; jj < 8; ++jj) {
      int row = mt * 16 + jj + rAdd;
      Tw[row * 256 + col] += oacc[t][jj] + b2v;
    }
  }
}

// ---------------------------------------------------------------------------
// Kernel: window reverse  T[w][i][d] -> out[b][d][r][c]
// ---------------------------------------------------------------------------
__global__ void winrev_k(const float* __restrict__ T, float* __restrict__ out) {
  int idx = blockIdx.x * 256 + threadIdx.x;     // 16,777,216
  int c = idx & 127, r = (idx >> 7) & 127;
  int b = idx >> 22;
  int wv = b * 256 + (r >> 3) * 16 + (c >> 3);
  int i = (r & 7) * 8 + (c & 7);
  int d = (idx >> 14) & 255;
  out[idx] = T[((size_t)wv * 64 + i) * 256 + d];
}

// ---------------------------------------------------------------------------
// Host-side orchestration
// ---------------------------------------------------------------------------
extern "C" void kernel_launch(void* const* d_in, const int* in_sizes, int n_in,
                              void* d_out, int out_size, void* d_ws, size_t ws_size,
                              hipStream_t stream) {
  (void)in_sizes; (void)n_in; (void)out_size; (void)ws_size;
  const float* x        = (const float*)d_in[0];
  const float* W_patch  = (const float*)d_in[1];
  const float* b_patch  = (const float*)d_in[2];
  const float* ln1_g    = (const float*)d_in[3];
  const float* ln1_b    = (const float*)d_in[4];
  const float* Wqkv     = (const float*)d_in[5];
  const float* headsita = (const float*)d_in[6];
  const float* bias_tbl = (const float*)d_in[7];
  const float* Wout     = (const float*)d_in[8];
  const float* bout     = (const float*)d_in[9];
  const float* ln2_g    = (const float*)d_in[10];
  const float* ln2_b    = (const float*)d_in[11];
  const float* W1       = (const float*)d_in[12];
  const float* b1       = (const float*)d_in[13];
  const float* W2       = (const float*)d_in[14];
  const float* b2       = (const float*)d_in[15];
  float* out = (float*)d_out;

  // packed-weight region offsets (in halves)
  constexpr size_t H_PATCH = 0;
  constexpr size_t H_QKV0  = H_PATCH + 65536;
  constexpr size_t H_QKV1  = H_QKV0 + 196608;
  constexpr size_t H_OUT0  = H_QKV1 + 196608;
  constexpr size_t H_OUT1  = H_OUT0 + 65536;
  constexpr size_t H_W10   = H_OUT1 + 65536;
  constexpr size_t H_W11   = H_W10 + 262144;
  constexpr size_t H_W20   = H_W11 + 262144;
  constexpr size_t H_W21   = H_W20 + 262144;
  constexpr size_t H_END   = H_W21 + 262144;          // 1,638,400 halves
  // byte offsets of fp32/f16 working buffers
  constexpr size_t B_T  = H_END * 2;                  // T: 1024*64*256 f32
  constexpr size_t B_Q  = B_T + 67108864;             // Qh f16
  constexpr size_t B_K  = B_Q + 33554432;             // Kh f16
  constexpr size_t B_V  = B_K + 33554432;             // Vt f16
  constexpr size_t B_O  = B_V + 33554432;             // O  f16
  constexpr size_t B_BP = B_O + 33554432;             // biaspos f32 (128KB)

  char* ws = (char*)d_ws;
  _Float16* pw   = (_Float16*)ws;
  float*    T    = (float*)(ws + B_T);
  _Float16* Qh   = (_Float16*)(ws + B_Q);
  _Float16* Kh   = (_Float16*)(ws + B_K);
  _Float16* Vt   = (_Float16*)(ws + B_V);
  _Float16* O    = (_Float16*)(ws + B_O);
  float*    BP   = (float*)(ws + B_BP);

  const size_t hQkv[2] = {H_QKV0, H_QKV1};
  const size_t hOut[2] = {H_OUT0, H_OUT1};
  const size_t hW1[2]  = {H_W10, H_W11};
  const size_t hW2[2]  = {H_W20, H_W21};

  // 1) pack all weights to f16 fragment layout (runs every call; idempotent)
  pack_w_k<<<16, 256, 0, stream>>>(W_patch, pw + H_PATCH, 256, 256);
  for (int l = 0; l < 2; ++l) {
    pack_w_k<<<48, 256, 0, stream>>>(Wqkv + (size_t)l * 196608, pw + hQkv[l], 256, 768);
    pack_w_k<<<16, 256, 0, stream>>>(Wout + (size_t)l * 65536,  pw + hOut[l], 256, 256);
    pack_w_k<<<64, 256, 0, stream>>>(W1   + (size_t)l * 262144, pw + hW1[l],  256, 1024);
    pack_w_k<<<64, 256, 0, stream>>>(W2   + (size_t)l * 262144, pw + hW2[l],  1024, 256);
  }

  // 2) window partition + patch embedding
  patch_embed_k<<<NWIN, 256, 0, stream>>>(x, pw + H_PATCH, b_patch, T);

  // 3) two transformer layers
  constexpr int SH_ATTN = 8 * 64 * 72 * 2;                         // 73,728 B
  constexpr int SH_MLP  = NTOK * LDSTR * 2 + NTOK * MIDSTR * 2 + 2048; // 167,936 B
  for (int l = 0; l < 2; ++l) {
    biaspos_k<<<128, 256, 0, stream>>>(bias_tbl + (size_t)l * 1800,
                                       headsita + (size_t)l * 8, BP);
    ln_qkv_k<<<NWIN, 256, 0, stream>>>(T, ln1_g + l * 256, ln1_b + l * 256,
                                       pw + hQkv[l], Qh, Kh, Vt);
    attn_k<<<NWIN, 256, SH_ATTN, stream>>>(
        Qh, Kh, Vt, BP, out + 16777216 + (size_t)l * 33554432, O);
    outproj_k<<<NWIN, 256, 0, stream>>>(O, pw + hOut[l], bout + l * 256, T);
    ln2_mlp_k<<<NWIN, 256, SH_MLP, stream>>>(
        T, ln2_g + l * 256, ln2_b + l * 256, pw + hW1[l], b1 + (size_t)l * 1024,
        pw + hW2[l], b2 + l * 256);
  }

  // 4) window reverse into NCHW output
  winrev_k<<<65536, 256, 0, stream>>>(T, out);
}
